// GaussKernelAttn_2499670966480
// MI455X (gfx1250) — matrix operand order repaired and checked by reading, not compiled, not run on hardware
//
#include <hip/hip_runtime.h>

#define B_   8
#define N_   1032
#define NP_  1056        // N padded to multiple of 32 (zeroed pad rows)
#define C_   768
#define H_   12
#define HD_  64
#define M_   (B_ * N_)   // 8256 tokens

typedef __attribute__((ext_vector_type(16))) __bf16 v16bf;
typedef __attribute__((ext_vector_type(8)))  float  v8f;

union Frag {
  v16bf v;
  uint4 q[2];
};

__device__ __forceinline__ unsigned short f2bf(float f) {
  unsigned u = __builtin_bit_cast(unsigned, f);
  unsigned r = ((u >> 16) & 1u) + 0x7FFFu;   // round-to-nearest-even
  return (unsigned short)((u + r) >> 16);
}

__device__ __forceinline__ v8f wmma_bf16(Frag a, Frag b, v8f c) {
  return __builtin_amdgcn_wmma_f32_16x16x32_bf16(
      /*neg_a=*/false, a.v, /*neg_b=*/false, b.v,
      /*c_mod=*/(short)0, c, /*reuse_a=*/false, /*reuse_b=*/false);
}

// Gaussian positional bias, separable, computed on the fly.
// Tokens 0..7 are prefix (bias 0); patch index p = r-8, grid 32x32.
__device__ __forceinline__ float gauss_add(int r, int c) {
  if (r < 8 || c < 8) return 0.0f;
  int ri = (r - 8) >> 5, rj = (r - 8) & 31;
  int ci = (c - 8) >> 5, cj = (c - 8) & 31;
  int di = ri - ci, dj = rj - cj;
  return __expf(-0.02f * (float)(di * di + dj * dj));  // 1/(2*5^2) = 0.02
}

// ---------------------------------------------------------------- converts
__global__ void cvt_f32_bf16(const float* __restrict__ in,
                             unsigned short* __restrict__ out, int n) {
  int i = blockIdx.x * blockDim.x + threadIdx.x;
  if (i < n) out[i] = f2bf(in[i]);
}

// zero the padded rows (n in [1032,1056)) of Kbuf and Vt
__global__ void zeropad(unsigned short* __restrict__ Kbuf,
                        unsigned short* __restrict__ Vt) {
  int idx = blockIdx.x * blockDim.x + threadIdx.x;
  const int tot = 96 * 24 * 64;
  if (idx < tot) {
    int d = idx & 63;
    int t = idx >> 6;            // bh*24 + nrel
    int nrel = t % 24, bh = t / 24;
    Kbuf[((size_t)bh * NP_ + 1032 + nrel) * HD_ + d] = 0;
  } else if (idx < 2 * tot) {
    int j = idx - tot;
    int nrel = j % 24;
    int t = j / 24;              // bh*64 + d
    int d = t & 63, bh = t >> 6;
    Vt[((size_t)bh * HD_ + d) * NP_ + 1032 + nrel] = 0;
  }
}

// ---------------------------------------------------------------- QKV GEMM
// Wave tile: 32 (M) x 64 (N). out[m,j] = sum_k x[m,k]*wkv[j,k] + b_qkv[768+j]
// j <  768 : K -> Kbuf[b,h,n,d]      (row-major per head, padded rows)
// j >= 768 : V -> Vt  [b,h,d,n]      (transposed, padded cols)
__global__ __launch_bounds__(128) void qkv_gemm(
    const unsigned short* __restrict__ xb,
    const unsigned short* __restrict__ wkv,
    const float* __restrict__ bqkv,
    unsigned short* __restrict__ Kbuf,
    unsigned short* __restrict__ Vt) {
  const int lane = threadIdx.x & 31;
  const int wv   = threadIdx.x >> 5;
  const int mt   = blockIdx.x * 4 + wv;     // 32-row tiles, 0..257
  if (mt >= M_ / 32) return;
  const int nb   = blockIdx.y * 64;
  const int ln16 = lane & 15, hi = lane >> 4;

  v8f zero = {0.f, 0.f, 0.f, 0.f, 0.f, 0.f, 0.f, 0.f};
  v8f acc[2][4];
#pragma unroll
  for (int s = 0; s < 2; ++s)
#pragma unroll
    for (int j = 0; j < 4; ++j) acc[s][j] = zero;

  const unsigned short* ar0 = xb + (size_t)(mt * 32 + ln16) * C_;
  const unsigned short* ar1 = ar0 + 16 * C_;
  const int abase = hi * 8;

  for (int kc = 0; kc < 24; ++kc) {
    const int k0 = kc * 32;
    Frag a0, a1;
    a0.q[0] = *(const uint4*)(ar0 + k0 + abase);
    a0.q[1] = *(const uint4*)(ar0 + k0 + abase + 16);
    a1.q[0] = *(const uint4*)(ar1 + k0 + abase);
    a1.q[1] = *(const uint4*)(ar1 + k0 + abase + 16);
#pragma unroll
    for (int j = 0; j < 4; ++j) {
      Frag b;
      const unsigned short* br =
          wkv + (size_t)(nb + j * 16 + ln16) * C_ + k0 + hi * 8;
      b.q[0] = *(const uint4*)(br);
      b.q[1] = *(const uint4*)(br + 16);
      acc[0][j] = wmma_bf16(a0, b, acc[0][j]);
      acc[1][j] = wmma_bf16(a1, b, acc[1][j]);
    }
  }

#pragma unroll
  for (int s = 0; s < 2; ++s) {
#pragma unroll
    for (int j = 0; j < 4; ++j) {
      int n = nb + j * 16 + ln16;
      float bias = bqkv[768 + n];
#pragma unroll
      for (int i = 0; i < 8; ++i) {
        int m = mt * 32 + s * 16 + hi * 8 + i;   // global token
        float val = acc[s][j][i] + bias;
        int bbi = m / N_, tok = m % N_;
        unsigned short bv = f2bf(val);
        if (n < C_) {
          int hh = n >> 6, d = n & 63;
          Kbuf[(((size_t)(bbi * H_ + hh)) * NP_ + tok) * HD_ + d] = bv;
        } else {
          int n2 = n - C_;
          int hh = n2 >> 6, d = n2 & 63;
          Vt[(((size_t)(bbi * H_ + hh)) * HD_ + d) * NP_ + tok] = bv;
        }
      }
    }
  }
}

// ---------------------------------------------------------------- attention
// Per wave: 16 query rows of one (b,h). Flash-style streaming over key
// chunks of 32 with online softmax; Gauss bias computed on the fly.
__global__ __launch_bounds__(128) void attn_kernel(
    const unsigned short* __restrict__ Kbuf,
    const unsigned short* __restrict__ Vt,
    unsigned short* __restrict__ ctx) {
  __shared__ __align__(16) unsigned short lp[4][16 * 48];  // P transpose staging
  const int lane = threadIdx.x & 31;
  const int wv   = threadIdx.x >> 5;
  const int qt   = blockIdx.x * 4 + wv;
  if (qt >= 66) return;                       // 66 tiles cover NP_=1056
  const int bh = blockIdx.y;                  // b*12 + h
  const int h  = bh % H_;
  const int bb = bh / H_;
  const int qBase = qt * 16;
  const int ln16 = lane & 15, hi = lane >> 4;

  const unsigned short* Kh = Kbuf + (size_t)bh * NP_ * HD_;
  const unsigned short* Vh = Vt   + (size_t)bh * HD_ * NP_;

  // A-fragments of K (query side), hd split into two K=32 chunks
  Frag aK0, aK1;
  {
    const unsigned short* p = Kh + (size_t)(qBase + ln16) * HD_;
    int base = hi * 16;
    aK0.q[0] = *(const uint4*)(p + base);
    aK0.q[1] = *(const uint4*)(p + base + 8);
    aK1.q[0] = *(const uint4*)(p + 32 + base);
    aK1.q[1] = *(const uint4*)(p + 32 + base + 8);
  }

  v8f zero = {0.f, 0.f, 0.f, 0.f, 0.f, 0.f, 0.f, 0.f};
  v8f o[4];
#pragma unroll
  for (int dt = 0; dt < 4; ++dt) o[dt] = zero;
  float mrow[8], lrow[8];
#pragma unroll
  for (int i = 0; i < 8; ++i) { mrow[i] = -1e30f; lrow[i] = 0.f; }

  for (int kt = 0; kt < NP_ / 32; ++kt) {
    const int kBase = kt * 32;
    // B-fragments of K^T: columns = key tokens (rows of Kh are contiguous)
    Frag b00, b01, b10, b11;
    {
      const unsigned short* p0 = Kh + (size_t)(kBase + ln16) * HD_;
      const unsigned short* p1 = Kh + (size_t)(kBase + 16 + ln16) * HD_;
      int koff = hi * 16;
      b00.q[0] = *(const uint4*)(p0 + koff);
      b00.q[1] = *(const uint4*)(p0 + koff + 8);
      b01.q[0] = *(const uint4*)(p0 + 32 + koff);
      b01.q[1] = *(const uint4*)(p0 + 32 + koff + 8);
      b10.q[0] = *(const uint4*)(p1 + koff);
      b10.q[1] = *(const uint4*)(p1 + koff + 8);
      b11.q[0] = *(const uint4*)(p1 + 32 + koff);
      b11.q[1] = *(const uint4*)(p1 + 32 + koff + 8);
    }

    v8f s0 = zero, s1 = zero;
    s0 = wmma_bf16(aK0, b00, s0);
    s0 = wmma_bf16(aK1, b01, s0);
    s1 = wmma_bf16(aK0, b10, s1);
    s1 = wmma_bf16(aK1, b11, s1);

    // scale + gauss bias + key mask
    const int c0 = kBase + ln16;
    const int c1 = c0 + 16;
#pragma unroll
    for (int i = 0; i < 8; ++i) {
      int r = qBase + hi * 8 + i;
      float v0 = s0[i] * 0.125f + gauss_add(r, c0);
      float v1 = s1[i] * 0.125f + gauss_add(r, c1);
      if (c0 >= N_) v0 = -1e30f;
      if (c1 >= N_) v1 = -1e30f;
      s0[i] = v0;
      s1[i] = v1;
    }

    // online softmax: row reductions across the 16 lanes owning each row
#pragma unroll
    for (int i = 0; i < 8; ++i) {
      float mx = fmaxf(s0[i], s1[i]);
      mx = fmaxf(mx, __shfl_xor(mx, 1, 32));
      mx = fmaxf(mx, __shfl_xor(mx, 2, 32));
      mx = fmaxf(mx, __shfl_xor(mx, 4, 32));
      mx = fmaxf(mx, __shfl_xor(mx, 8, 32));
      float nm = fmaxf(mrow[i], mx);
      float sf = __expf(mrow[i] - nm);
      float p0 = __expf(s0[i] - nm);
      float p1 = __expf(s1[i] - nm);
      float rs = p0 + p1;
      rs += __shfl_xor(rs, 1, 32);
      rs += __shfl_xor(rs, 2, 32);
      rs += __shfl_xor(rs, 4, 32);
      rs += __shfl_xor(rs, 8, 32);
      lrow[i] = lrow[i] * sf + rs;
      mrow[i] = nm;
      o[0][i] *= sf; o[1][i] *= sf; o[2][i] *= sf; o[3][i] *= sf;
      s0[i] = p0;
      s1[i] = p1;
    }

    // P (C-layout) -> LDS -> A-fragment (row-major transpose)
    unsigned short* L = lp[wv];
#pragma unroll
    for (int i = 0; i < 8; ++i) {
      int row = hi * 8 + i;
      L[row * 48 + ln16]      = f2bf(s0[i]);
      L[row * 48 + 16 + ln16] = f2bf(s1[i]);
    }
    asm volatile("" ::: "memory");   // keep DS store->load order (HW is in-order)
    Frag pf;
    {
      const unsigned short* lr = L + ln16 * 48;
      int base = hi * 8;
      pf.q[0] = *(const uint4*)(lr + base);
      pf.q[1] = *(const uint4*)(lr + base + 16);
    }
    asm volatile("" ::: "memory");

    // P @ V : reduction over the 32 keys of this chunk, Vt rows contiguous
#pragma unroll
    for (int dt = 0; dt < 4; ++dt) {
      Frag bv;
      const unsigned short* pv =
          Vh + (size_t)(dt * 16 + ln16) * NP_ + kBase + hi * 8;
      bv.q[0] = *(const uint4*)(pv);
      bv.q[1] = *(const uint4*)(pv + 16);
      o[dt] = wmma_bf16(pf, bv, o[dt]);
    }
  }

  // normalize + store context (bf16), masked to valid tokens
#pragma unroll
  for (int i = 0; i < 8; ++i) {
    int r = qBase + hi * 8 + i;
    if (r < N_) {
      float inv = 1.0f / lrow[i];
      size_t ob = ((size_t)(bb * N_ + r)) * C_ + h * HD_ + ln16;
      ctx[ob]      = f2bf(o[0][i] * inv);
      ctx[ob + 16] = f2bf(o[1][i] * inv);
      ctx[ob + 32] = f2bf(o[2][i] * inv);
      ctx[ob + 48] = f2bf(o[3][i] * inv);
    }
  }
}

// ---------------------------------------------------------------- proj GEMM
// Wave tile: 32 (M) x 64 (N), f32 output with bias.
__global__ __launch_bounds__(128) void proj_gemm(
    const unsigned short* __restrict__ ctx,
    const unsigned short* __restrict__ wp,
    const float* __restrict__ bp,
    float* __restrict__ out) {
  const int lane = threadIdx.x & 31;
  const int wv   = threadIdx.x >> 5;
  const int mt   = blockIdx.x * 4 + wv;     // 32-row tiles, 0..257
  if (mt >= M_ / 32) return;
  const int nb   = blockIdx.y * 64;
  const int ln16 = lane & 15, hi = lane >> 4;

  v8f zero = {0.f, 0.f, 0.f, 0.f, 0.f, 0.f, 0.f, 0.f};
  v8f acc[2][4];
#pragma unroll
  for (int s = 0; s < 2; ++s)
#pragma unroll
    for (int j = 0; j < 4; ++j) acc[s][j] = zero;

  const unsigned short* ar0 = ctx + (size_t)(mt * 32 + ln16) * C_;
  const unsigned short* ar1 = ar0 + 16 * C_;
  const int abase = hi * 8;

  for (int kc = 0; kc < 24; ++kc) {
    const int k0 = kc * 32;
    Frag a0, a1;
    a0.q[0] = *(const uint4*)(ar0 + k0 + abase);
    a0.q[1] = *(const uint4*)(ar0 + k0 + abase + 16);
    a1.q[0] = *(const uint4*)(ar1 + k0 + abase);
    a1.q[1] = *(const uint4*)(ar1 + k0 + abase + 16);
#pragma unroll
    for (int j = 0; j < 4; ++j) {
      Frag b;
      const unsigned short* br =
          wp + (size_t)(nb + j * 16 + ln16) * C_ + k0 + hi * 8;
      b.q[0] = *(const uint4*)(br);
      b.q[1] = *(const uint4*)(br + 16);
      acc[0][j] = wmma_bf16(a0, b, acc[0][j]);
      acc[1][j] = wmma_bf16(a1, b, acc[1][j]);
    }
  }

#pragma unroll
  for (int s = 0; s < 2; ++s) {
#pragma unroll
    for (int j = 0; j < 4; ++j) {
      int n = nb + j * 16 + ln16;
      float bias = bp[n];
#pragma unroll
      for (int i = 0; i < 8; ++i) {
        int m = mt * 32 + s * 16 + hi * 8 + i;
        out[(size_t)m * C_ + n] = acc[s][j][i] + bias;
      }
    }
  }
}

// ---------------------------------------------------------------- launch
extern "C" void kernel_launch(void* const* d_in, const int* in_sizes, int n_in,
                              void* d_out, int out_size, void* d_ws, size_t ws_size,
                              hipStream_t stream) {
  (void)in_sizes; (void)n_in; (void)out_size; (void)ws_size;
  const float* x      = (const float*)d_in[0];
  const float* w_qkv  = (const float*)d_in[1];
  const float* b_qkv  = (const float*)d_in[2];
  const float* w_proj = (const float*)d_in[3];
  const float* b_proj = (const float*)d_in[4];
  float* out = (float*)d_out;

  char* ws = (char*)d_ws;
  size_t off = 0;
  auto alloc = [&](size_t bytes) -> void* {
    void* p = ws + off;
    off = (off + bytes + 255) & ~(size_t)255;
    return p;
  };
  unsigned short* xb   = (unsigned short*)alloc((size_t)M_ * C_ * 2);
  unsigned short* wkv  = (unsigned short*)alloc((size_t)2 * C_ * C_ * 2);
  unsigned short* wpj  = (unsigned short*)alloc((size_t)C_ * C_ * 2);
  unsigned short* Kbuf = (unsigned short*)alloc((size_t)B_ * H_ * NP_ * HD_ * 2);
  unsigned short* Vt   = (unsigned short*)alloc((size_t)B_ * H_ * HD_ * NP_ * 2);
  unsigned short* ctx  = (unsigned short*)alloc((size_t)M_ * C_ * 2);

  int nx = M_ * C_;
  cvt_f32_bf16<<<(nx + 255) / 256, 256, 0, stream>>>(x, xb, nx);
  int nwkv = 2 * C_ * C_;
  cvt_f32_bf16<<<(nwkv + 255) / 256, 256, 0, stream>>>(w_qkv + (size_t)C_ * C_, wkv, nwkv);
  int nwp = C_ * C_;
  cvt_f32_bf16<<<(nwp + 255) / 256, 256, 0, stream>>>(w_proj, wpj, nwp);
  int npad = 2 * 96 * 24 * 64;
  zeropad<<<(npad + 255) / 256, 256, 0, stream>>>(Kbuf, Vt);

  qkv_gemm<<<dim3(65, 24), 128, 0, stream>>>(xb, wkv, b_qkv, Kbuf, Vt);
  attn_kernel<<<dim3(17, 96), 128, 0, stream>>>(Kbuf, Vt, ctx);
  proj_gemm<<<dim3(65, 12), 128, 0, stream>>>(ctx, wpj, b_proj, out);
}
